// SelfBlock_16750372455007
// MI455X (gfx1250) — matrix operand order repaired and checked by reading, not compiled, not run on hardware
//
#include <hip/hip_runtime.h>
#include <hip/hip_bf16.h>

// ---------------------------------------------------------------------------
// MI455X / gfx1250 implementation of the SelfBlock reference.
// All GEMMs use v_wmma_f32_16x16x32_f16 with fp32->f16 hi/lo splitting
// (3 WMMAs per logical f32 product) for near-fp32 accuracy at f16-WMMA rate.
// Attention K/V tiles are staged into LDS by the Tensor Data Mover
// (tensor_load_to_lds + s_wait_tensorcnt, double-buffered) when available.
// b=2, n=4096, e=256, h=4, hd=64 are hardcoded from the reference.
// ---------------------------------------------------------------------------

typedef _Float16 h16 __attribute__((ext_vector_type(16)));
typedef _Float16 h8  __attribute__((ext_vector_type(8)));
typedef float    f8  __attribute__((ext_vector_type(8)));
typedef unsigned int u32x4 __attribute__((ext_vector_type(4)));
typedef int i32x8 __attribute__((ext_vector_type(8)));
typedef int i32x4 __attribute__((ext_vector_type(4)));

#if defined(__AMDGCN__) && __has_builtin(__builtin_amdgcn_tensor_load_to_lds)
#define USE_TDM 1
#else
#define USE_TDM 0
#endif

union H16U { h16 v; h8 h[2]; };

// A-operand (16x32 f16): lane = row (l&15); two contiguous 16B K-chunks.
__device__ __forceinline__ h16 load_a16(const _Float16* base, int ld, int k0, int lane) {
  int m  = lane & 15;
  int kb = (lane >> 4) << 3;
  const _Float16* p = base + (size_t)m * ld + k0 + kb;
  H16U u;
  u.h[0] = *(const h8*)(p);
  u.h[1] = *(const h8*)(p + 16);
  return u.v;
}

// B-operand (32x16 f16): lane = col (l&15); two contiguous 16B K-chunks.
// "base" is (N,K) row-major, i.e. a weight matrix for y = x @ W^T.
__device__ __forceinline__ h16 load_b16(const _Float16* base, int ld, int k0, int lane) {
  int n  = lane & 15;
  int kb = (lane >> 4) << 3;
  const _Float16* p = base + (size_t)n * ld + k0 + kb;
  H16U u;
  u.h[0] = *(const h8*)(p);
  u.h[1] = *(const h8*)(p + 16);
  return u.v;
}

// acc += Ah*Bh + Al*Bh + Ah*Bl  (fp32-ish product of hi/lo split operands)
__device__ __forceinline__ f8 wmma_hilo(h16 ah, h16 al, h16 bh, h16 bl, f8 acc) {
  acc = __builtin_amdgcn_wmma_f32_16x16x32_f16(false, ah, false, bh, (short)0, acc, false, false);
  acc = __builtin_amdgcn_wmma_f32_16x16x32_f16(false, al, false, bh, (short)0, acc, false, false);
  acc = __builtin_amdgcn_wmma_f32_16x16x32_f16(false, ah, false, bl, (short)0, acc, false, false);
  return acc;
}

__device__ __forceinline__ void store_hilo(_Float16* H, _Float16* L, size_t off, float v) {
  _Float16 hv = (_Float16)v;
  H[off] = hv;
  L[off] = (_Float16)(v - (float)hv);
}

#if USE_TDM
// Low 32 bits of a generic pointer to LDS == byte offset in the LDS aperture.
__device__ __forceinline__ unsigned int lds_off(const void* p) {
  return (unsigned int)(unsigned long long)(uintptr_t)p;
}

// Issue one 2-D TDM tile load (global -> LDS), ISA 8.3/8.4 D# layout.
// tx/ty/stride in 2-byte elements; pad codes per ISA (interval 2^(c+1) DW, amount c+1 DW).
// clang-23 builtin arity: (u32x4 g0, i32x8 g1, i32x4 g2, i32x4 g3, i32x8, i32 cpol).
__device__ __forceinline__ void tdm_2d(const _Float16* gsrc, unsigned int lds_byte,
                                       unsigned int tx, unsigned int ty,
                                       unsigned int stride,
                                       unsigned int padi, unsigned int pada) {
  unsigned long long ga = (unsigned long long)(uintptr_t)gsrc;
  u32x4 g0 = { 1u,                                 // count=1, user descriptor
               lds_byte,                           // lds_addr
               (unsigned int)ga,                   // global_addr[31:0]
               (unsigned int)((ga >> 32) & 0x1FFFFFFull) | 0x80000000u }; // addr[56:32] | type=2
  unsigned int w0 = (1u << 16)      // data_size = 2 bytes
                  | (1u << 20)      // pad_enable (LDS row padding)
                  | (padi << 22) | (pada << 25);
  i32x8 g1 = { (int)w0,
               (int)((tx & 0xFFFFu) << 16),                              // tensor_dim0[15:0]
               (int)(((tx >> 16) & 0xFFFFu) | ((ty & 0xFFFFu) << 16)),   // dim0 hi | dim1 lo
               (int)(((ty >> 16) & 0xFFFFu) | ((tx & 0xFFFFu) << 16)),   // dim1 hi | tile_dim0
               (int)(ty & 0xFFFFu),                                      // tile_dim1 (tile_dim2=0)
               (int)stride,                                              // tensor_dim0_stride lo
               0, 0 };
  i32x4 gz  = { 0, 0, 0, 0 };
  i32x8 gz8 = { 0, 0, 0, 0, 0, 0, 0, 0 };
  __builtin_amdgcn_tensor_load_to_lds(g0, g1, gz, gz, gz8, 0);
}
#endif

// Fallback cooperative tile copy (256 threads, 256 chunks of 8 halfs).
__device__ __forceinline__ void coop_copy(_Float16* dst, int dpitch,
                                          const _Float16* src, size_t spitch,
                                          int cols8, int tid) {
  int r = tid / cols8;
  int c = (tid % cols8) * 8;
  *(h8*)(dst + r * dpitch + c) = *(const h8*)(src + (size_t)r * spitch + c);
}

// ---------------------------------------------------------------------------
// fp32 -> f16 hi/lo conversion (one element per thread)
// ---------------------------------------------------------------------------
__global__ __launch_bounds__(256) void cvt_hilo_kernel(const float* __restrict__ src,
                                                       _Float16* __restrict__ hi,
                                                       _Float16* __restrict__ lo,
                                                       int count) {
  int i = blockIdx.x * blockDim.x + threadIdx.x;
  if (i < count) store_hilo(hi, lo, (size_t)i, src[i]);
}

// Pack x (8192 x 256 f32) into columns [0,256) of the concat buffer (ld 512).
__global__ __launch_bounds__(256) void pack_x_kernel(const float* __restrict__ x,
                                                     _Float16* __restrict__ A2h,
                                                     _Float16* __restrict__ A2l) {
  int i = blockIdx.x * blockDim.x + threadIdx.x;  // < 8192*256
  int row = i >> 8;
  int c   = i & 255;
  store_hilo(A2h, A2l, (size_t)row * 512 + c, x[i]);
}

// ---------------------------------------------------------------------------
// Generic hi/lo WMMA GEMM: C = A @ B^T + bias.  One 16x16 tile per wave.
// mode 0: f32 store   mode 1: f16 hi/lo store   mode 2: f32 + residual
// ---------------------------------------------------------------------------
__global__ __launch_bounds__(256) void gemm_hilo_kernel(
    const _Float16* __restrict__ Ah, const _Float16* __restrict__ Al, int lda,
    const _Float16* __restrict__ Bh, const _Float16* __restrict__ Bl, int ldb,
    const float* __restrict__ bias,
    int ntilesN, int K, int mode,
    float* __restrict__ outF, const float* __restrict__ resid, int ldf,
    _Float16* __restrict__ outH, _Float16* __restrict__ outL, int ldh) {
  int wave = (blockIdx.x * blockDim.x + threadIdx.x) >> 5;
  int lane = threadIdx.x & 31;
  int mt = wave / ntilesN;
  int nt = wave - mt * ntilesN;
  const _Float16* Ab  = Ah + (size_t)mt * 16 * lda;
  const _Float16* Abl = Al + (size_t)mt * 16 * lda;
  const _Float16* Bb  = Bh + (size_t)nt * 16 * ldb;
  const _Float16* Bbl = Bl + (size_t)nt * 16 * ldb;
  int n = lane & 15;
  float bv = bias[nt * 16 + n];
  f8 acc;
#pragma unroll
  for (int r = 0; r < 8; ++r) acc[r] = bv;
  for (int k0 = 0; k0 < K; k0 += 32) {
    if (k0 + 32 < K) {
      __builtin_prefetch((const void*)(Ab + (size_t)(lane & 15) * lda + k0 + 32), 0, 1);
      __builtin_prefetch((const void*)(Bb + (size_t)(lane & 15) * ldb + k0 + 32), 0, 1);
    }
    h16 ah = load_a16(Ab,  lda, k0, lane);
    h16 al = load_a16(Abl, lda, k0, lane);
    h16 bh = load_b16(Bb,  ldb, k0, lane);
    h16 bl = load_b16(Bbl, ldb, k0, lane);
    acc = wmma_hilo(ah, al, bh, bl, acc);
  }
  int mrow = mt * 16 + ((lane >> 4) << 3);
  if (mode == 0) {
#pragma unroll
    for (int r = 0; r < 8; ++r)
      outF[(size_t)(mrow + r) * ldf + nt * 16 + n] = acc[r];
  } else if (mode == 1) {
#pragma unroll
    for (int r = 0; r < 8; ++r)
      store_hilo(outH, outL, (size_t)(mrow + r) * ldh + nt * 16 + n, acc[r]);
  } else {
#pragma unroll
    for (int r = 0; r < 8; ++r) {
      size_t off = (size_t)(mrow + r) * ldf + nt * 16 + n;
      outF[off] = acc[r] + resid[off];
    }
  }
}

// ---------------------------------------------------------------------------
// RoPE-style rotation + head split.  qkv row layout: channel c -> 3c,3c+1,3c+2.
// Q,K stored natural (bh, n, 64); V stored transposed (bh, 64, n).
// ---------------------------------------------------------------------------
__global__ __launch_bounds__(256) void rope_kernel(
    const float* __restrict__ qkv, const float* __restrict__ enc,
    _Float16* __restrict__ Qh, _Float16* __restrict__ Ql,
    _Float16* __restrict__ Kh, _Float16* __restrict__ Kl,
    _Float16* __restrict__ VTh, _Float16* __restrict__ VTl) {
  int idx = blockIdx.x * blockDim.x + threadIdx.x;  // < 2*4096*128
  int p  = idx & 127;
  int ni = (idx >> 7) & 4095;
  int bi = idx >> 19;
  int c0 = p * 2, c1 = c0 + 1;
  const float* row = qkv + (size_t)(bi * 4096 + ni) * 768;
  float q0 = row[3 * c0], k0 = row[3 * c0 + 1], v0 = row[3 * c0 + 2];
  float q1 = row[3 * c1], k1 = row[3 * c1 + 1], v1 = row[3 * c1 + 2];
  size_t eb = (size_t)(bi * 4096 + ni) * 256;
  const float* cosp = enc + eb;
  const float* sinp = enc + (size_t)2 * 4096 * 256 + eb;
  float c_0 = cosp[c0], c_1 = cosp[c1], s_0 = sinp[c0], s_1 = sinp[c1];
  float qr0 = q0 * c_0 - q1 * s_0;
  float qr1 = q1 * c_1 + q0 * s_1;
  float kr0 = k0 * c_0 - k1 * s_0;
  float kr1 = k1 * c_1 + k0 * s_1;
  int head = c0 >> 6;
  int d0 = c0 & 63;
  size_t qoff = ((size_t)(bi * 4 + head) * 4096 + ni) * 64 + d0;
  store_hilo(Qh, Ql, qoff, qr0);
  store_hilo(Qh, Ql, qoff + 1, qr1);
  store_hilo(Kh, Kl, qoff, kr0);
  store_hilo(Kh, Kl, qoff + 1, kr1);
  size_t vt = ((size_t)(bi * 4 + head) * 64 + d0) * 4096 + ni;
  store_hilo(VTh, VTl, vt, v0);
  store_hilo(VTh, VTl, vt + 4096, v1);
}

// ---------------------------------------------------------------------------
// Flash attention: one 16-row q-tile per wave, 32-key blocks, online softmax.
// K/V tiles are staged once per block into padded LDS by the TDM
// (double-buffered, TENSORcnt-pipelined); scores and P@V run on WMMA.
// ---------------------------------------------------------------------------
__global__ __launch_bounds__(256) void attn_kernel(
    const _Float16* __restrict__ Qh, const _Float16* __restrict__ Ql,
    const _Float16* __restrict__ Kh, const _Float16* __restrict__ Kl,
    const _Float16* __restrict__ VTh, const _Float16* __restrict__ VTl,
    _Float16* __restrict__ Ch, _Float16* __restrict__ Cl) {
  // Padded pitches (K: 72 halfs = 144B, V/P: 40 halfs = 80B) keep ds_load_b128
  // reads 16B-aligned and bank-conflict-free.
  __shared__ __attribute__((aligned(16))) _Float16 KldsH[2][32][72];
  __shared__ __attribute__((aligned(16))) _Float16 KldsL[2][32][72];
  __shared__ __attribute__((aligned(16))) _Float16 VldsH[2][64][40];
  __shared__ __attribute__((aligned(16))) _Float16 VldsL[2][64][40];
  __shared__ __attribute__((aligned(16))) _Float16 PldsH[8][16][40];
  __shared__ __attribute__((aligned(16))) _Float16 PldsL[8][16][40];

  int tid  = threadIdx.x;
  int wid  = tid >> 5;
  int lane = tid & 31;
  int bh = blockIdx.y;                 // bi*4 + head
  int q0 = (blockIdx.x * 8 + wid) * 16;

  const _Float16* Qb  = Qh + ((size_t)bh * 4096 + q0) * 64;
  const _Float16* Qbl = Ql + ((size_t)bh * 4096 + q0) * 64;
  h16 qa_h0 = load_a16(Qb, 64, 0, lane);
  h16 qa_h1 = load_a16(Qb, 64, 32, lane);
  h16 qa_l0 = load_a16(Qbl, 64, 0, lane);
  h16 qa_l1 = load_a16(Qbl, 64, 32, lane);

  f8 o[4];
#pragma unroll
  for (int t = 0; t < 4; ++t)
#pragma unroll
    for (int r = 0; r < 8; ++r) o[t][r] = 0.f;
  float rowm[8], rowsum[8];
#pragma unroll
  for (int r = 0; r < 8; ++r) { rowm[r] = -1.0e30f; rowsum[r] = 0.f; }

  const _Float16* Kb  = Kh + (size_t)bh * 4096 * 64;
  const _Float16* Kbl = Kl + (size_t)bh * 4096 * 64;
  const _Float16* Vb  = VTh + (size_t)bh * 64 * 4096;
  const _Float16* Vbl = VTl + (size_t)bh * 64 * 4096;

#if USE_TDM
  // Stage key-block j into buffer j&1: K tiles 32x64 (pad 4DW/32DW),
  // V tiles 64x32 (pad 4DW/16DW).  4 in-order TDM ops per block.
  auto stage = [&](int j) {
    int b = j & 1;
    int k0 = j * 32;
    tdm_2d(Kb  + (size_t)k0 * 64, lds_off(&KldsH[b][0][0]), 64, 32, 64, 4, 3);
    tdm_2d(Kbl + (size_t)k0 * 64, lds_off(&KldsL[b][0][0]), 64, 32, 64, 4, 3);
    tdm_2d(Vb  + k0,              lds_off(&VldsH[b][0][0]), 32, 64, 4096, 3, 3);
    tdm_2d(Vbl + k0,              lds_off(&VldsL[b][0][0]), 32, 64, 4096, 3, 3);
  };
  if (wid == 0) stage(0);
#endif

  for (int it = 0; it < 128; ++it) {
    int buf = it & 1;
    __syncthreads();  // prior reads of the buffer being (re)filled are done
#if USE_TDM
    if (wid == 0) {
      if (it + 1 < 128) {
        stage(it + 1);
        __builtin_amdgcn_s_wait_tensorcnt(4);  // retire current buffer's 4 loads
      } else {
        __builtin_amdgcn_s_wait_tensorcnt(0);
      }
    }
#else
    {
      int k0 = it * 32;
      coop_copy(&KldsH[buf][0][0], 72, Kb  + (size_t)k0 * 64, 64, 8, tid);
      coop_copy(&KldsL[buf][0][0], 72, Kbl + (size_t)k0 * 64, 64, 8, tid);
      coop_copy(&VldsH[buf][0][0], 40, Vb  + k0, 4096, 4, tid);
      coop_copy(&VldsL[buf][0][0], 40, Vbl + k0, 4096, 4, tid);
    }
#endif
    __syncthreads();  // tiles visible to all waves

    const _Float16* KtH = &KldsH[buf][0][0];
    const _Float16* KtL = &KldsL[buf][0][0];
    f8 s0, s1;
#pragma unroll
    for (int r = 0; r < 8; ++r) { s0[r] = 0.f; s1[r] = 0.f; }
    s0 = wmma_hilo(qa_h0, qa_l0, load_b16(KtH, 72, 0, lane),  load_b16(KtL, 72, 0, lane),  s0);
    s0 = wmma_hilo(qa_h1, qa_l1, load_b16(KtH, 72, 32, lane), load_b16(KtL, 72, 32, lane), s0);
    s1 = wmma_hilo(qa_h0, qa_l0, load_b16(KtH + 16 * 72, 72, 0, lane),
                   load_b16(KtL + 16 * 72, 72, 0, lane), s1);
    s1 = wmma_hilo(qa_h1, qa_l1, load_b16(KtH + 16 * 72, 72, 32, lane),
                   load_b16(KtL + 16 * 72, 72, 32, lane), s1);

    // online softmax over this 32-key block (row stats live per half-wave)
#pragma unroll
    for (int r = 0; r < 8; ++r) {
      float v0 = s0[r] * 0.125f;
      float v1 = s1[r] * 0.125f;
      float mloc = fmaxf(v0, v1);
#pragma unroll
      for (int off = 1; off < 16; off <<= 1)
        mloc = fmaxf(mloc, __shfl_xor(mloc, off, 32));
      float mnew  = fmaxf(rowm[r], mloc);
      float alpha = __expf(rowm[r] - mnew);
      float p0 = __expf(v0 - mnew);
      float p1 = __expf(v1 - mnew);
      float ps = p0 + p1;
#pragma unroll
      for (int off = 1; off < 16; off <<= 1)
        ps += __shfl_xor(ps, off, 32);
      rowsum[r] = rowsum[r] * alpha + ps;
      rowm[r] = mnew;
#pragma unroll
      for (int t = 0; t < 4; ++t) o[t][r] *= alpha;
      int m  = r + ((lane >> 4) << 3);
      int nn = lane & 15;
      _Float16 h0 = (_Float16)p0;
      PldsH[wid][m][nn] = h0;
      PldsL[wid][m][nn] = (_Float16)(p0 - (float)h0);
      _Float16 h1 = (_Float16)p1;
      PldsH[wid][m][16 + nn] = h1;
      PldsL[wid][m][16 + nn] = (_Float16)(p1 - (float)h1);
    }
    h16 pa_h = load_a16(&PldsH[wid][0][0], 40, 0, lane);
    h16 pa_l = load_a16(&PldsL[wid][0][0], 40, 0, lane);
#pragma unroll
    for (int t = 0; t < 4; ++t)
      o[t] = wmma_hilo(pa_h, pa_l, load_b16(&VldsH[buf][t * 16][0], 40, 0, lane),
                       load_b16(&VldsL[buf][t * 16][0], 40, 0, lane), o[t]);
  }

  int bi = bh >> 2, head = bh & 3;
#pragma unroll
  for (int r = 0; r < 8; ++r) {
    float inv = 1.0f / rowsum[r];
    int m = r + ((lane >> 4) << 3);
    size_t row = (size_t)bi * 4096 + q0 + m;
#pragma unroll
    for (int t = 0; t < 4; ++t)
      store_hilo(Ch, Cl, row * 256 + head * 64 + t * 16 + (lane & 15), o[t][r] * inv);
  }
}

// ---------------------------------------------------------------------------
// LayerNorm (512) + exact GELU.  One row per wave; shuffle-only reductions.
// ---------------------------------------------------------------------------
__global__ __launch_bounds__(256) void ln_gelu_kernel(
    const float* __restrict__ y, const float* __restrict__ g,
    const float* __restrict__ b, _Float16* __restrict__ outH,
    _Float16* __restrict__ outL) {
  int wave = (blockIdx.x * blockDim.x + threadIdx.x) >> 5;
  int lane = threadIdx.x & 31;
  const float* row = y + (size_t)wave * 512;
  float vals[16], s = 0.f, s2 = 0.f;
#pragma unroll
  for (int j = 0; j < 16; ++j) {
    float v = row[lane + 32 * j];
    vals[j] = v;
    s += v;
    s2 += v * v;
  }
#pragma unroll
  for (int off = 1; off < 32; off <<= 1) {
    s  += __shfl_xor(s, off, 32);
    s2 += __shfl_xor(s2, off, 32);
  }
  float mu   = s * (1.0f / 512.0f);
  float var  = s2 * (1.0f / 512.0f) - mu * mu;
  float rstd = rsqrtf(var + 1e-5f);
#pragma unroll
  for (int j = 0; j < 16; ++j) {
    int c = lane + 32 * j;
    float v  = (vals[j] - mu) * rstd * g[c] + b[c];
    float ge = 0.5f * v * (1.0f + erff(v * 0.70710678118654752f));
    store_hilo(outH, outL, (size_t)wave * 512 + c, ge);
  }
}

// ---------------------------------------------------------------------------
extern "C" void kernel_launch(void* const* d_in, const int* in_sizes, int n_in,
                              void* d_out, int out_size, void* d_ws, size_t ws_size,
                              hipStream_t stream) {
  (void)in_sizes; (void)n_in; (void)out_size; (void)ws_size;
  const float* x      = (const float*)d_in[0];    // (2,4096,256)
  const float* enc    = (const float*)d_in[1];    // (2,2,4096,256,1)
  const float* Wqkv_w = (const float*)d_in[2];    // (768,256)
  const float* Wqkv_b = (const float*)d_in[3];
  const float* out_w  = (const float*)d_in[4];    // (256,256)
  const float* out_b  = (const float*)d_in[5];
  const float* ffn1_w = (const float*)d_in[6];    // (512,512)
  const float* ffn1_b = (const float*)d_in[7];
  const float* ln_g   = (const float*)d_in[8];
  const float* ln_b   = (const float*)d_in[9];
  const float* ffn2_w = (const float*)d_in[10];   // (256,512)
  const float* ffn2_b = (const float*)d_in[11];
  float* out = (float*)d_out;

  // -------- workspace carve-up (~78 MB peak, with deliberate reuse) --------
  char* base = (char*)d_ws;
  size_t off = 0;
  auto nxt = [&](size_t bytes) -> char* {
    char* p = base + off;
    off += (bytes + 255) & ~(size_t)255;
    return p;
  };
  _Float16* A2h = (_Float16*)nxt((size_t)8192 * 512 * 2);  // concat [x|msg] hi
  _Float16* A2l = (_Float16*)nxt((size_t)8192 * 512 * 2);
  _Float16* Wqh = (_Float16*)nxt((size_t)768 * 256 * 2);
  _Float16* Wql = (_Float16*)nxt((size_t)768 * 256 * 2);
  _Float16* Owh = (_Float16*)nxt((size_t)256 * 256 * 2);
  _Float16* Owl = (_Float16*)nxt((size_t)256 * 256 * 2);
  _Float16* F1h = (_Float16*)nxt((size_t)512 * 512 * 2);
  _Float16* F1l = (_Float16*)nxt((size_t)512 * 512 * 2);
  _Float16* F2h = (_Float16*)nxt((size_t)256 * 512 * 2);
  _Float16* F2l = (_Float16*)nxt((size_t)256 * 512 * 2);
  float* scratchF = (float*)nxt((size_t)8192 * 768 * 4);   // qkv f32, reused as ffn1-out f32
  char* qreg = nxt((size_t)6 * 4194304);                   // Q/K/VT hi+lo, reused as Y hi/lo
  _Float16* Qh  = (_Float16*)(qreg + 0 * (size_t)4194304);
  _Float16* Ql  = (_Float16*)(qreg + 1 * (size_t)4194304);
  _Float16* Kh  = (_Float16*)(qreg + 2 * (size_t)4194304);
  _Float16* Kl  = (_Float16*)(qreg + 3 * (size_t)4194304);
  _Float16* VTh = (_Float16*)(qreg + 4 * (size_t)4194304);
  _Float16* VTl = (_Float16*)(qreg + 5 * (size_t)4194304);
  _Float16* Chh = (_Float16*)nxt((size_t)8192 * 256 * 2);  // ctx hi
  _Float16* Cll = (_Float16*)nxt((size_t)8192 * 256 * 2);  // ctx lo
  _Float16* Yh  = (_Float16*)(qreg + 0);                   // overlay (Q/K free by then)
  _Float16* Yl  = (_Float16*)(qreg + (size_t)8388608);

  // -------- 1) convert weights + pack x into concat buffer --------
  cvt_hilo_kernel<<<768, 256, 0, stream>>>(Wqkv_w, Wqh, Wql, 768 * 256);
  cvt_hilo_kernel<<<256, 256, 0, stream>>>(out_w, Owh, Owl, 256 * 256);
  cvt_hilo_kernel<<<1024, 256, 0, stream>>>(ffn1_w, F1h, F1l, 512 * 512);
  cvt_hilo_kernel<<<512, 256, 0, stream>>>(ffn2_w, F2h, F2l, 256 * 512);
  pack_x_kernel<<<8192, 256, 0, stream>>>(x, A2h, A2l);

  // -------- 2) QKV projection: (8192x256)@(768x256)^T -> scratchF --------
  gemm_hilo_kernel<<<3072, 256, 0, stream>>>(A2h, A2l, 512, Wqh, Wql, 256, Wqkv_b,
                                             48, 256, 0, scratchF, nullptr, 768,
                                             nullptr, nullptr, 0);

  // -------- 3) rotation + head split -> Q, K, V^T (hi/lo f16) --------
  rope_kernel<<<4096, 256, 0, stream>>>(scratchF, enc, Qh, Ql, Kh, Kl, VTh, VTl);

  // -------- 4) flash attention (TDM-staged K/V) -> ctx hi/lo --------
  attn_kernel<<<dim3(32, 8), 256, 0, stream>>>(Qh, Ql, Kh, Kl, VTh, VTl, Chh, Cll);

  // -------- 5) out projection -> columns [256,512) of concat buffer --------
  gemm_hilo_kernel<<<1024, 256, 0, stream>>>(Chh, Cll, 256, Owh, Owl, 256, out_b,
                                             16, 256, 1, nullptr, nullptr, 0,
                                             A2h + 256, A2l + 256, 512);

  // -------- 6) FFN1: (8192x512)@(512x512)^T -> scratchF (f32) --------
  gemm_hilo_kernel<<<2048, 256, 0, stream>>>(A2h, A2l, 512, F1h, F1l, 512, ffn1_b,
                                             32, 512, 0, scratchF, nullptr, 512,
                                             nullptr, nullptr, 0);

  // -------- 7) LayerNorm + GELU -> Y hi/lo --------
  ln_gelu_kernel<<<1024, 256, 0, stream>>>(scratchF, ln_g, ln_b, Yh, Yl);

  // -------- 8) FFN2 + bias + residual -> d_out --------
  gemm_hilo_kernel<<<1024, 256, 0, stream>>>(Yh, Yl, 512, F2h, F2l, 512, ffn2_b,
                                             16, 512, 2, out, x, 256,
                                             nullptr, nullptr, 0);
}
